// AttentionOp_60988535603899
// MI455X (gfx1250) — compile-verified, matrix operand order checked by reading, not run on hardware
//
#include <hip/hip_runtime.h>

// ---------------------------------------------------------------------------
// Types for CDNA5 WMMA
// ---------------------------------------------------------------------------
typedef __attribute__((ext_vector_type(16))) __bf16 v16bf;
typedef __attribute__((ext_vector_type(8)))  float  v8f;
typedef __attribute__((ext_vector_type(2)))  float  v2f;
typedef __attribute__((ext_vector_type(2)))  __bf16 v2bf;
typedef int v4i __attribute__((vector_size(16)));   // matches async builtin sig

union FragBF { v16bf v; uint4 u[2]; };

// native bf16 converts (v_cvt_pk_bf16_f32 on gfx1250)
__device__ __forceinline__ unsigned pack2(float lo, float hi) {
  v2f f = {lo, hi};
  v2bf h = __builtin_convertvector(f, v2bf);
  return __builtin_bit_cast(unsigned, h);
}
__device__ __forceinline__ unsigned short f2bf(float f) {
  return __builtin_bit_cast(unsigned short, (__bf16)f);
}
__device__ __forceinline__ float bf2f(unsigned short h) {
  return __uint_as_float(((unsigned)h) << 16);
}

// CDNA5 async global->LDS copy (GLOBAL_LOAD_ASYNC_TO_LDS_B128, ASYNCcnt).
// Signature (from hipcc diagnostic): (AS1 v4i*, AS3 v4i*, imm offset, imm cpol)
// Per ISA 08: INST_OFFSET is added to BOTH the global and LDS addresses.
#define ASYNC_CP16(gp, lp, OFF)                                       \
  __builtin_amdgcn_global_load_async_to_lds_b128(                     \
      (__attribute__((address_space(1))) v4i*)(gp),                   \
      (__attribute__((address_space(3))) v4i*)(lp), (OFF), 0)

__device__ __forceinline__ void wait_async0() {
  asm volatile("s_wait_asynccnt 0" ::: "memory");
}

// ---------------------------------------------------------------------------
// f32 -> bf16 convert (vector x4)
// ---------------------------------------------------------------------------
__global__ void cvt_f32_bf16(const float* __restrict__ in,
                             unsigned short* __restrict__ out, int n4) {
  int i = blockIdx.x * blockDim.x + threadIdx.x;
  if (i < n4) {
    float4 f = ((const float4*)in)[i];
    uint2 o;
    o.x = pack2(f.x, f.y);
    o.y = pack2(f.z, f.w);
    ((uint2*)out)[i] = o;
  }
}

// ---------------------------------------------------------------------------
// Tiled bf16 WMMA GEMM:  C(MxN) = A(MxK) * B(NxK)^T
//   A: f32 (A_F32=true, converted in-register) or bf16 (async-LDS path)
//   B: bf16 row-major (N,K) [weights layout], always async-LDS
//   Optional f32 C, optional bf16 C copy, optional f32 residual added.
// Block: 256 threads (8 waves), tile 128x128, K-step 32, double-buffered LDS.
// ---------------------------------------------------------------------------
#define LDS_STRIDE 40   // ushorts per row: 32 data + 8 pad = 80B, 16B aligned

template <bool A_F32>
__global__ __launch_bounds__(256) void gemm_bf16(
    const void* __restrict__ Aptr, const unsigned short* __restrict__ Bmat,
    float* __restrict__ C, unsigned short* __restrict__ Cbf,
    const float* __restrict__ residual, int M, int N, int K) {
  __shared__ unsigned short As[2][128 * LDS_STRIDE];
  __shared__ unsigned short Bs[2][128 * LDS_STRIDE];

  const int t = threadIdx.x;
  const int m0 = blockIdx.y * 128;
  const int n0 = blockIdx.x * 128;

  // tile loader mapping: 2 threads per row, 16 elems each
  const int ldrow = t >> 1;
  const int ldcol = (t & 1) << 4;
  const int aoff = ldrow * LDS_STRIDE + ldcol;

  // wave/fragment mapping (per ISA 16-bit A layout: lanes 0-15 K 0-7/16-23,
  // lanes 16-31 K 8-15/24-31; B symmetric with lane = N)
  const int lane = t & 31;
  const int wave = t >> 5;
  const int wm = wave >> 2;             // 0..1  -> 64 rows
  const int wn = wave & 3;              // 0..3  -> 32 cols
  const int fr = lane & 15;
  const int kb = (lane >> 4) << 3;      // 0 or 8

  const float* Af = (const float*)Aptr;
  const unsigned short* Ah = (const unsigned short*)Aptr;

  v8f acc[4][2];
#pragma unroll
  for (int i = 0; i < 4; ++i)
#pragma unroll
    for (int j = 0; j < 2; ++j) {
      v8f z = {0.f, 0.f, 0.f, 0.f, 0.f, 0.f, 0.f, 0.f};
      acc[i][j] = z;
    }

  // one K-step of WMMAs out of LDS buffer `cur`
  auto mma_step = [&](int cur) {
    FragBF fb[2];
#pragma unroll
    for (int j = 0; j < 2; ++j) {
      const unsigned short* p =
          &Bs[cur][(wn * 32 + j * 16 + fr) * LDS_STRIDE + kb];
      fb[j].u[0] = *(const uint4*)(p + 0);
      fb[j].u[1] = *(const uint4*)(p + 16);
    }
#pragma unroll
    for (int i = 0; i < 4; ++i) {
      FragBF fa;
      const unsigned short* p =
          &As[cur][(wm * 64 + i * 16 + fr) * LDS_STRIDE + kb];
      fa.u[0] = *(const uint4*)(p + 0);
      fa.u[1] = *(const uint4*)(p + 16);
#pragma unroll
      for (int j = 0; j < 2; ++j) {
        acc[i][j] = __builtin_amdgcn_wmma_f32_16x16x32_bf16(
            false, fa.v, false, fb[j].v, (short)0, acc[i][j], false, false);
      }
    }
  };

  const int KT = K >> 5;
  const unsigned short* Bg = Bmat + (size_t)(n0 + ldrow) * K + ldcol;

  if (!A_F32) {
    // ---------------- fully-async double-buffered pipeline ----------------
    const unsigned short* Ag = Ah + (size_t)(m0 + ldrow) * K + ldcol;

    ASYNC_CP16(Ag, &As[0][aoff], 0);
    ASYNC_CP16(Ag, &As[0][aoff], 16);
    ASYNC_CP16(Bg, &Bs[0][aoff], 0);
    ASYNC_CP16(Bg, &Bs[0][aoff], 16);
    wait_async0();
    __syncthreads();

    for (int kt = 0; kt < KT; ++kt) {
      const int cur = kt & 1;
      const bool more = (kt + 1) < KT;
      if (more) {
        const unsigned short* a = Ag + (size_t)(kt + 1) * 32;
        const unsigned short* b = Bg + (size_t)(kt + 1) * 32;
        ASYNC_CP16(a, &As[1 - cur][aoff], 0);
        ASYNC_CP16(a, &As[1 - cur][aoff], 16);
        ASYNC_CP16(b, &Bs[1 - cur][aoff], 0);
        ASYNC_CP16(b, &Bs[1 - cur][aoff], 16);
      }
      mma_step(cur);
      if (more) {
        wait_async0();
        __syncthreads();
      }
    }
  } else {
    // ------- f32 A: register-staged convert; B via async-LDS -------------
    const float* Ag = Af + (size_t)(m0 + ldrow) * K + ldcol;

    float4 a0 = *(const float4*)(Ag + 0);
    float4 a1 = *(const float4*)(Ag + 4);
    float4 a2 = *(const float4*)(Ag + 8);
    float4 a3 = *(const float4*)(Ag + 12);
    ASYNC_CP16(Bg, &Bs[0][aoff], 0);
    ASYNC_CP16(Bg, &Bs[0][aoff], 16);
    {
      uint4 sa0, sa1;
      sa0.x = pack2(a0.x, a0.y); sa0.y = pack2(a0.z, a0.w);
      sa0.z = pack2(a1.x, a1.y); sa0.w = pack2(a1.z, a1.w);
      sa1.x = pack2(a2.x, a2.y); sa1.y = pack2(a2.z, a2.w);
      sa1.z = pack2(a3.x, a3.y); sa1.w = pack2(a3.z, a3.w);
      *(uint4*)&As[0][aoff + 0] = sa0;
      *(uint4*)&As[0][aoff + 8] = sa1;
    }
    wait_async0();
    __syncthreads();

    for (int kt = 0; kt < KT; ++kt) {
      const int cur = kt & 1;
      const bool more = (kt + 1) < KT;
      if (more) {
        const float* p = Ag + (size_t)(kt + 1) * 32;
        a0 = *(const float4*)(p + 0);
        a1 = *(const float4*)(p + 4);
        a2 = *(const float4*)(p + 8);
        a3 = *(const float4*)(p + 12);
        if (kt + 2 < KT)
          __builtin_prefetch(Ag + (size_t)(kt + 2) * 32, 0, 0);
      }
      mma_step(cur);
      if (more) {
        __syncthreads();
        const int nxt = (kt + 1) & 1;
        const unsigned short* b = Bg + (size_t)(kt + 1) * 32;
        ASYNC_CP16(b, &Bs[nxt][aoff], 0);
        ASYNC_CP16(b, &Bs[nxt][aoff], 16);
        uint4 sa0, sa1;
        sa0.x = pack2(a0.x, a0.y); sa0.y = pack2(a0.z, a0.w);
        sa0.z = pack2(a1.x, a1.y); sa0.w = pack2(a1.z, a1.w);
        sa1.x = pack2(a2.x, a2.y); sa1.y = pack2(a2.z, a2.w);
        sa1.z = pack2(a3.x, a3.y); sa1.w = pack2(a3.z, a3.w);
        *(uint4*)&As[nxt][aoff + 0] = sa0;
        *(uint4*)&As[nxt][aoff + 8] = sa1;
        wait_async0();
        __syncthreads();
      }
    }
  }

  // ---- epilogue: C/D layout -> VGPR r: M=r (+8 for hi lanes), N=lane&15 ----
#pragma unroll
  for (int i = 0; i < 4; ++i) {
#pragma unroll
    for (int j = 0; j < 2; ++j) {
      const int mBase = m0 + wm * 64 + i * 16 + ((lane >> 4) << 3);
      const int nIdx = n0 + wn * 32 + j * 16 + fr;
#pragma unroll
      for (int r = 0; r < 8; ++r) {
        const size_t idx = (size_t)(mBase + r) * (size_t)N + (size_t)nIdx;
        float v = acc[i][j][r];
        if (residual) v += residual[idx];
        if (C) C[idx] = v;
        if (Cbf) Cbf[idx] = f2bf(v);
      }
    }
  }
}

// ---------------------------------------------------------------------------
// Linear attention: kv[bh][d][m] = sum_l (elu(k)+1)[l,d] * v[l,m]
//                   ksum[bh][d]  = sum_l (elu(k)+1)[l,d]
// ---------------------------------------------------------------------------
__global__ __launch_bounds__(256) void kv_kernel(
    const unsigned short* __restrict__ qkv, float* __restrict__ kv,
    float* __restrict__ ksum) {
  __shared__ float sk[32][64];
  __shared__ float sv[32][64];

  const int bh = blockIdx.x;          // 0..63
  const int b = bh >> 4, h = bh & 15;
  const int t = threadIdx.x;

  const int d_own = t >> 2;           // 0..63
  const int mBase = (t & 3) * 16;     // 16 m-cells per thread

  float acc[16];
#pragma unroll
  for (int i = 0; i < 16; ++i) acc[i] = 0.f;
  float ks = 0.f;

  const size_t rowBase = (size_t)b * 4096 * 3072 + (size_t)h * 64;

  for (int l0 = 0; l0 < 4096; l0 += 32) {
    __syncthreads();
    for (int e = t; e < 2048; e += 256) {
      const int l = e >> 6, d = e & 63;
      const size_t r = rowBase + (size_t)(l0 + l) * 3072;
      const float kvl = bf2f(qkv[r + 1024 + d]);
      sk[l][d] = kvl > 0.f ? kvl + 1.f : __expf(kvl);   // elu(k)+1
      sv[l][d] = bf2f(qkv[r + 2048 + d]);
    }
    __syncthreads();
#pragma unroll 4
    for (int l = 0; l < 32; ++l) {
      const float kd = sk[l][d_own];
#pragma unroll
      for (int mm = 0; mm < 16; ++mm) acc[mm] += kd * sv[l][mBase + mm];
      if ((t & 3) == 0) ks += kd;
    }
  }

  const size_t ob = ((size_t)bh * 64 + d_own) * 64 + mBase;
#pragma unroll
  for (int mm = 0; mm < 16; ++mm) kv[ob + mm] = acc[mm];
  if ((t & 3) == 0) ksum[(size_t)bh * 64 + d_own] = ks;
}

// ---------------------------------------------------------------------------
// out[bh][l][m] = (qf . kv[:,m]) / (qf . ksum + 1e-8), bf16 store with the
// reference's raw (B,H,L,D)->(B,L,H*D) reshape folded into the index.
// ---------------------------------------------------------------------------
__global__ __launch_bounds__(256) void attn_out_kernel(
    const unsigned short* __restrict__ qkv, const float* __restrict__ kv,
    const float* __restrict__ ksum, unsigned short* __restrict__ outbf) {
  __shared__ float skv[64 * 64];
  __shared__ float sks[64];

  const int bh = blockIdx.x >> 4;     // 0..63
  const int b = bh >> 4, h = bh & 15;
  const int t = threadIdx.x;

  for (int e = t; e < 4096; e += 256) skv[e] = kv[(size_t)bh * 4096 + e];
  if (t < 64) sks[t] = ksum[(size_t)bh * 64 + t];
  __syncthreads();

  const int l = (blockIdx.x & 15) * 256 + t;
  const size_t qoff = ((size_t)(b * 4096 + l)) * 3072 + (size_t)h * 64;

  float qf[64];
  const uint4* qp = (const uint4*)(qkv + qoff);
#pragma unroll
  for (int c = 0; c < 8; ++c) {
    uint4 u = qp[c];
    const unsigned ww[4] = {u.x, u.y, u.z, u.w};
#pragma unroll
    for (int j = 0; j < 4; ++j) {
      float lo = bf2f((unsigned short)(ww[j] & 0xffffu));
      float hi = bf2f((unsigned short)(ww[j] >> 16));
      qf[c * 8 + j * 2 + 0] = lo > 0.f ? lo + 1.f : __expf(lo);
      qf[c * 8 + j * 2 + 1] = hi > 0.f ? hi + 1.f : __expf(hi);
    }
  }

  float norm = 0.f;
#pragma unroll
  for (int d = 0; d < 64; ++d) norm += qf[d] * sks[d];
  const float inv = 1.f / (norm + 1e-8f);

  // reshape: flat = (h*L + l)*D + m ; row' = flat>>10 ; e = flat&1023
  const size_t flat = ((size_t)h * 4096 + (size_t)l) * 64;
  const size_t obase = ((size_t)b * 4096 + (flat >> 10)) * 1024 + (flat & 1023);
  unsigned* op = (unsigned*)(outbf + obase);   // 64-elem aligned

  for (int m = 0; m < 64; m += 2) {
    float o0 = 0.f, o1 = 0.f;
#pragma unroll
    for (int d = 0; d < 64; ++d) {
      o0 += qf[d] * skv[d * 64 + m + 0];
      o1 += qf[d] * skv[d * 64 + m + 1];
    }
    op[m >> 1] = pack2(o0 * inv, o1 * inv);
  }
}

// ---------------------------------------------------------------------------
// In-place RMSNorm over 1024 cols: y = y * rsqrt(mean(y*y)+eps) * norm_w
// ---------------------------------------------------------------------------
__global__ __launch_bounds__(256) void rms_kernel(float* __restrict__ y,
                                                  const float* __restrict__ w) {
  __shared__ float red[256];
  const int row = blockIdx.x;
  const int t = threadIdx.x;
  float* yr = y + (size_t)row * 1024;

  float s = 0.f;
#pragma unroll
  for (int c = t; c < 1024; c += 256) { float v = yr[c]; s += v * v; }
  red[t] = s;
  __syncthreads();
  for (int off = 128; off > 0; off >>= 1) {
    if (t < off) red[t] += red[t + off];
    __syncthreads();
  }
  const float rms = rsqrtf(red[0] * (1.f / 1024.f) + 1.1920928955078125e-7f);
#pragma unroll
  for (int c = t; c < 1024; c += 256) yr[c] = yr[c] * rms * w[c];
}

// ---------------------------------------------------------------------------
// Host launch
// ---------------------------------------------------------------------------
extern "C" void kernel_launch(void* const* d_in, const int* in_sizes, int n_in,
                              void* d_out, int out_size, void* d_ws,
                              size_t ws_size, hipStream_t stream) {
  const float* x = (const float*)d_in[0];       // (4,4096,512)
  const float* w_in = (const float*)d_in[1];    // (1024,512)
  const float* w_qkv = (const float*)d_in[2];   // (3072,1024)
  const float* w_out = (const float*)d_in[3];   // (1024,1024)
  const float* norm_w = (const float*)d_in[4];  // (1024,)

  const int M = 16384;       // B*L
  const int LAT = 1024;      // latent
  const int KIN = 512;
  const int NQKV = 3072;

  char* ws = (char*)d_ws;
  size_t off = 0;
  auto carve = [&](size_t bytes) {
    char* p = ws + off;
    off = (off + bytes + 255) & ~(size_t)255;
    return p;
  };
  unsigned short* w_in_bf = (unsigned short*)carve((size_t)LAT * KIN * 2);
  unsigned short* w_qkv_bf = (unsigned short*)carve((size_t)NQKV * LAT * 2);
  unsigned short* w_out_bf = (unsigned short*)carve((size_t)LAT * LAT * 2);
  float* xproj_f = (float*)carve((size_t)M * LAT * 4);
  unsigned short* xproj_bf = (unsigned short*)carve((size_t)M * LAT * 2);
  unsigned short* qkv_bf = (unsigned short*)carve((size_t)M * NQKV * 2);
  float* kv_ws = (float*)carve((size_t)64 * 64 * 64 * 4);
  float* ksum_ws = (float*)carve((size_t)64 * 64 * 4);
  unsigned short* attn_bf = (unsigned short*)carve((size_t)M * LAT * 2);
  (void)ws_size;

  // 1) weights -> bf16
  {
    int n4;
    n4 = (LAT * KIN) / 4;
    cvt_f32_bf16<<<(n4 + 255) / 256, 256, 0, stream>>>(w_in, w_in_bf, n4);
    n4 = (NQKV * LAT) / 4;
    cvt_f32_bf16<<<(n4 + 255) / 256, 256, 0, stream>>>(w_qkv, w_qkv_bf, n4);
    n4 = (LAT * LAT) / 4;
    cvt_f32_bf16<<<(n4 + 255) / 256, 256, 0, stream>>>(w_out, w_out_bf, n4);
  }

  // 2) x_proj = x @ w_in^T   (f32 A -> f32 C + bf16 copy)
  {
    dim3 grid(LAT / 128, M / 128);
    gemm_bf16<true><<<grid, 256, 0, stream>>>(x, w_in_bf, xproj_f, xproj_bf,
                                              (const float*)nullptr, M, LAT,
                                              KIN);
  }

  // 3) qkv = x_proj @ w_qkv^T  (bf16 out only)
  {
    dim3 grid(NQKV / 128, M / 128);
    gemm_bf16<false><<<grid, 256, 0, stream>>>(
        xproj_bf, w_qkv_bf, (float*)nullptr, qkv_bf, (const float*)nullptr, M,
        NQKV, LAT);
  }

  // 4) kv + ksum per (b,h)
  kv_kernel<<<64, 256, 0, stream>>>(qkv_bf, kv_ws, ksum_ws);

  // 5) attention out (with faithful reshape) -> bf16
  attn_out_kernel<<<1024, 256, 0, stream>>>(qkv_bf, kv_ws, ksum_ws, attn_bf);

  // 6) y = attn_out @ w_out^T + x_proj   (into d_out)
  {
    dim3 grid(LAT / 128, M / 128);
    gemm_bf16<false><<<grid, 256, 0, stream>>>(attn_bf, w_out_bf,
                                               (float*)d_out,
                                               (unsigned short*)nullptr,
                                               xproj_f, M, LAT, LAT);
  }

  // 7) in-place RMSNorm
  rms_kernel<<<M, 256, 0, stream>>>((float*)d_out, norm_w);
}
